// MultiHeadAttention_21242908246413
// MI455X (gfx1250) — matrix operand order repaired and checked by reading, not compiled, not run on hardware
//
#include <hip/hip_runtime.h>
#include <hip/hip_bf16.h>
#include <math.h>

// ---------------------------------------------------------------------------
// MHA (B=2,S=2048,D=1024,H=16,dq=64) for gfx1250:
//   f16 WMMA GEMMs + TDM-fed flash attention + fused residual + LayerNorm.
// ---------------------------------------------------------------------------

typedef __attribute__((ext_vector_type(16))) _Float16 v16h;
typedef __attribute__((ext_vector_type(8)))  _Float16 h8;
typedef __attribute__((ext_vector_type(8)))  float    v8f;
typedef __attribute__((ext_vector_type(4)))  unsigned int u32x4;
typedef __attribute__((ext_vector_type(8)))  unsigned int u32x8;

union F16x16 { v16h v; h8 h2[2]; };

constexpr int kB   = 2;
constexpr int kS   = 2048;
constexpr int kD   = 1024;
constexpr int kH   = 16;
constexpr int kDq  = 64;
constexpr int kTok = kB * kS;

// ---------------------------------------------------------------- conversion
__global__ void cvt_f32_f16(const float* __restrict__ src,
                            _Float16* __restrict__ dst, int n) {
  int i = blockIdx.x * blockDim.x + threadIdx.x;
  if (i < n) dst[i] = (_Float16)src[i];
}

// ------------------------------------------------------------------- GEMM
// C = A(MxK,f16) * Bw(KxN,f16); either f16 store (Ch) or f32 store + residual
// (Cf/resid). 128x128 tile, 8 waves, each wave 32x64 (2x4 wmma tiles).
__global__ __launch_bounds__(256) void gemm_f16_wmma(
    const _Float16* __restrict__ A,
    const _Float16* __restrict__ Bw,
    _Float16* __restrict__ Ch,
    float* __restrict__ Cf,
    const float* __restrict__ resid,
    int M, int N, int K)
{
  __shared__ __align__(16) _Float16 sA[128 * 32];   // [m][k]
  __shared__ __align__(16) _Float16 sBT[128 * 32];  // [n][k] (transposed)

  const int tid  = threadIdx.x;
  const int lane = tid & 31, wave = tid >> 5;
  const int hlf  = lane >> 4, l16 = lane & 15;
  const int wm   = wave >> 1;   // 0..3 -> 32-row group
  const int wn   = wave & 1;    // 0..1 -> 64-col group
  const int m0   = blockIdx.y * 128;
  const int n0   = blockIdx.x * 128;

  v8f zero = {};
  v8f acc[2][4];
  for (int i = 0; i < 2; ++i)
    for (int j = 0; j < 4; ++j) acc[i][j] = zero;

  for (int k0 = 0; k0 < K; k0 += 32) {
    if (k0 + 32 < K)  // next-slab speculative prefetch -> global_prefetch_b8
      __builtin_prefetch(&A[(size_t)(m0 + (tid >> 1)) * K + k0 + 32], 0, 1);

    // A tile: 128x32 f16 = 512 x 16B chunks, b128 in -> b128 out
    for (int ch = tid; ch < 512; ch += 256) {
      int r = ch >> 2, co = (ch & 3) * 8;
      *(h8*)&sA[r * 32 + co] =
          *(const h8*)&A[(size_t)(m0 + r) * K + (k0 + co)];
    }
    // B tile (transposed store): b128 global reads, 8-way LDS scatter
    for (int ch = tid; ch < 512; ch += 256) {
      int kk = ch >> 4, nc = (ch & 15) * 8;
      h8 val = *(const h8*)&Bw[(size_t)(k0 + kk) * N + (n0 + nc)];
      for (int j = 0; j < 8; ++j) sBT[(nc + j) * 32 + kk] = val[j];
    }
    __syncthreads();

    // A fragments: lane row M = l16; K = 8*hlf + e  and 16 + 8*hlf + e
    F16x16 af[2];
    for (int i = 0; i < 2; ++i) {
      const _Float16* base = &sA[(wm * 32 + i * 16 + l16) * 32 + hlf * 8];
      af[i].h2[0] = *(const h8*)base;
      af[i].h2[1] = *(const h8*)(base + 16);
    }
    // B fragments: lane col N = l16; K = 8*hlf + e  and 16 + 8*hlf + e
    F16x16 bf[4];
    for (int nt = 0; nt < 4; ++nt) {
      const _Float16* base = &sBT[(wn * 64 + nt * 16 + l16) * 32 + hlf * 8];
      bf[nt].h2[0] = *(const h8*)base;
      bf[nt].h2[1] = *(const h8*)(base + 16);
    }
    for (int i = 0; i < 2; ++i)
      for (int nt = 0; nt < 4; ++nt)
        acc[i][nt] = __builtin_amdgcn_wmma_f32_16x16x32_f16(
            false, af[i].v, false, bf[nt].v, (short)0, acc[i][nt],
            false, false);
    __syncthreads();
  }

  // C layout: VGPR r -> row r + 8*hlf ; lane -> col l16
  for (int i = 0; i < 2; ++i)
    for (int nt = 0; nt < 4; ++nt) {
      int col = n0 + wn * 64 + nt * 16 + l16;
      for (int r = 0; r < 8; ++r) {
        int row = m0 + wm * 32 + i * 16 + r + 8 * hlf;
        size_t off = (size_t)row * N + col;
        float v = acc[i][nt][r];
        if (Cf) Cf[off] = v + (resid ? resid[off] : 0.0f);
        else    Ch[off] = (_Float16)v;
      }
    }
}

// ------------------------------------------------------------ flash attention
// grid (B*H, S/128); 8 waves/block, 16 queries/wave; 32-key KV chunks in LDS.
// K tile is streamed by the Tensor Data Mover; V tile is transposed by VALU.
__global__ __launch_bounds__(256) void flash_attn_wmma(
    const _Float16* __restrict__ Qh,
    const _Float16* __restrict__ Kh,
    const _Float16* __restrict__ Vh,
    _Float16* __restrict__ Oh)
{
  __shared__ __align__(16) _Float16 sK[32 * 64];    // [key][dim] (TDM target)
  __shared__ __align__(16) _Float16 sVT[64 * 32];   // [dim][key]
  __shared__ __align__(16) _Float16 sP[8][16 * 32]; // per-wave P scratch

  const int tid  = threadIdx.x;
  const int lane = tid & 31, wave = tid >> 5;
  const int hlf  = lane >> 4, l16 = lane & 15;
  const int bh   = blockIdx.x;
  const int b    = bh / kH, h = bh % kH;
  const int tok0 = b * kS + blockIdx.y * 128 + wave * 16;
  const int dcol = h * kDq;

  // Q fragments (two K=32 slabs of the 64-wide head dim)
  F16x16 qf[2];
  for (int c = 0; c < 2; ++c) {
    const _Float16* qp = &Qh[(size_t)(tok0 + l16) * kD + dcol + c * 32 + hlf * 8];
    qf[c].h2[0] = *(const h8*)qp;
    qf[c].h2[1] = *(const h8*)(qp + 16);
  }

  v8f zero = {};
  v8f acc[4];
  for (int i = 0; i < 4; ++i) acc[i] = zero;
  float mrow[8], lrow[8];
  for (int r = 0; r < 8; ++r) { mrow[r] = -1e30f; lrow[r] = 0.0f; }
  const float sscale = 0.125f;  // 1/sqrt(64)

  const unsigned ldsK = (unsigned)(unsigned long long)(void*)&sK[0];

  for (int c0 = 0; c0 < kS; c0 += 32) {
    const size_t kvbase = (size_t)(b * kS + c0) * kD + dcol;

    // --- TDM: DMA the 32x64 f16 K tile (row stride D) into sK -------------
    if (wave == 0) {
      unsigned long long gaddr =
          (unsigned long long)(const void*)(Kh + kvbase);
      u32x4 g0;
      g0[0] = 1u;                                    // count=1, user mode
      g0[1] = ldsK;                                  // lds_addr
      g0[2] = (unsigned)(gaddr & 0xFFFFFFFFu);       // global_addr[31:0]
      g0[3] = ((unsigned)((gaddr >> 32) & 0x1FFFFFFu)) | 0x80000000u; // +type=2
      u32x8 g1;
      g1[0] = 0x00010000u;          // wg_mask=0, data_size=1 (2B)
      g1[1] = (64u << 16);          // tensor_dim0 = 64 elems
      g1[2] = ((unsigned)kS) << 16; // tensor_dim1 = S rows
      g1[3] = (64u << 16);          // tile_dim0 = 64
      g1[4] = 32u;                  // tile_dim1 = 32, tile_dim2 = 0
      g1[5] = (unsigned)kD;         // tensor_dim0_stride = 1024
      g1[6] = 0u;
      g1[7] = 0u;
      asm volatile("tensor_load_to_lds %0, %1" :: "s"(g0), "s"(g1) : "memory");
    }

    // --- VALU path: V tile, transposed into sVT (b128 global reads) -------
    {
      int ch = tid;  // 256 chunks of 8 f16
      int key = ch >> 3, dco = (ch & 7) * 8;
      h8 vv = *(const h8*)&Vh[kvbase + (size_t)key * kD + dco];
      for (int j = 0; j < 8; ++j) sVT[(dco + j) * 32 + key] = vv[j];
    }

    if (wave == 0) __builtin_amdgcn_s_wait_tensorcnt(0);
    __syncthreads();

    // scores: two 16-key tiles, each = 2 WMMAs over the 64 head dims
    v8f sc[2];
    for (int kt = 0; kt < 2; ++kt) {
      F16x16 kb0, kb1;
      const _Float16* kp = &sK[(kt * 16 + l16) * 64 + hlf * 8];
      kb0.h2[0] = *(const h8*)kp;        kb0.h2[1] = *(const h8*)(kp + 16);
      kb1.h2[0] = *(const h8*)(kp + 32); kb1.h2[1] = *(const h8*)(kp + 48);
      v8f s = zero;
      s = __builtin_amdgcn_wmma_f32_16x16x32_f16(false, qf[0].v, false, kb0.v,
                                                 (short)0, s, false, false);
      s = __builtin_amdgcn_wmma_f32_16x16x32_f16(false, qf[1].v, false, kb1.v,
                                                 (short)0, s, false, false);
      sc[kt] = s;
    }

    // online softmax: row r+8*hlf lives on 16 lanes of this half
    for (int r = 0; r < 8; ++r) {
      float v0 = sc[0][r] * sscale, v1 = sc[1][r] * sscale;
      float cm = fmaxf(v0, v1);
      for (int off = 1; off < 16; off <<= 1)
        cm = fmaxf(cm, __shfl_xor(cm, off, 32));
      float mnew   = fmaxf(mrow[r], cm);
      float sc_old = __expf(mrow[r] - mnew);
      float p0 = __expf(v0 - mnew), p1 = __expf(v1 - mnew);
      float ps = p0 + p1;
      for (int off = 1; off < 16; off <<= 1)
        ps += __shfl_xor(ps, off, 32);
      lrow[r] = lrow[r] * sc_old + ps;
      mrow[r] = mnew;
      for (int nt = 0; nt < 4; ++nt) acc[nt][r] = acc[nt][r] * sc_old;
      sP[wave][(r + 8 * hlf) * 32 + l16]      = (_Float16)p0;
      sP[wave][(r + 8 * hlf) * 32 + 16 + l16] = (_Float16)p1;
    }

    // P (16x32) as A fragment; V chunk as 4 B fragments; acc += P*V
    F16x16 pf;
    const _Float16* pp = &sP[wave][l16 * 32 + hlf * 8];
    pf.h2[0] = *(const h8*)pp;
    pf.h2[1] = *(const h8*)(pp + 16);
    for (int nt = 0; nt < 4; ++nt) {
      F16x16 vb;
      const _Float16* vp = &sVT[(nt * 16 + l16) * 32 + hlf * 8];
      vb.h2[0] = *(const h8*)vp;
      vb.h2[1] = *(const h8*)(vp + 16);
      acc[nt] = __builtin_amdgcn_wmma_f32_16x16x32_f16(
          false, pf.v, false, vb.v, (short)0, acc[nt], false, false);
    }
    __syncthreads();
  }

  for (int nt = 0; nt < 4; ++nt) {
    int col = dcol + nt * 16 + l16;
    for (int r = 0; r < 8; ++r) {
      int tok = tok0 + r + 8 * hlf;
      Oh[(size_t)tok * kD + col] = (_Float16)(acc[nt][r] / lrow[r]);
    }
  }
}

// ---------------------------------------------------------------- LayerNorm
__global__ __launch_bounds__(256) void layernorm_rows(
    const float* __restrict__ Y, float* __restrict__ out)
{
  __shared__ float red[256];
  const int row = blockIdx.x;
  const float* y = Y + (size_t)row * kD;
  float s = 0.0f;
  for (int i = threadIdx.x; i < kD; i += 256) s += y[i];
  red[threadIdx.x] = s; __syncthreads();
  for (int st = 128; st > 0; st >>= 1) {
    if (threadIdx.x < st) red[threadIdx.x] += red[threadIdx.x + st];
    __syncthreads();
  }
  float mu = red[0] * (1.0f / kD);
  __syncthreads();
  float v = 0.0f;
  for (int i = threadIdx.x; i < kD; i += 256) {
    float d = y[i] - mu; v += d * d;
  }
  red[threadIdx.x] = v; __syncthreads();
  for (int st = 128; st > 0; st >>= 1) {
    if (threadIdx.x < st) red[threadIdx.x] += red[threadIdx.x + st];
    __syncthreads();
  }
  float rstd = rsqrtf(red[0] * (1.0f / kD) + 1e-5f);
  for (int i = threadIdx.x; i < kD; i += 256)
    out[(size_t)row * kD + i] = (y[i] - mu) * rstd;
}

// ----------------------------------------------------------------- launcher
extern "C" void kernel_launch(void* const* d_in, const int* in_sizes, int n_in,
                              void* d_out, int out_size, void* d_ws, size_t ws_size,
                              hipStream_t stream) {
  (void)in_sizes; (void)n_in; (void)out_size; (void)ws_size;
  const float* ipt = (const float*)d_in[0];
  const float* Wq  = (const float*)d_in[1];
  const float* Wk  = (const float*)d_in[2];
  const float* Wv  = (const float*)d_in[3];
  const float* Wo  = (const float*)d_in[4];
  float* out = (float*)d_out;

  char* ws = (char*)d_ws;
  _Float16* Xh  = (_Float16*)ws; ws += (size_t)kTok * kD * sizeof(_Float16);
  _Float16* Wqh = (_Float16*)ws; ws += (size_t)kD * kD * sizeof(_Float16);
  _Float16* Wkh = (_Float16*)ws; ws += (size_t)kD * kD * sizeof(_Float16);
  _Float16* Wvh = (_Float16*)ws; ws += (size_t)kD * kD * sizeof(_Float16);
  _Float16* Woh = (_Float16*)ws; ws += (size_t)kD * kD * sizeof(_Float16);
  _Float16* Qh  = (_Float16*)ws; ws += (size_t)kTok * kD * sizeof(_Float16);
  _Float16* Kh  = (_Float16*)ws; ws += (size_t)kTok * kD * sizeof(_Float16);
  _Float16* Vh  = (_Float16*)ws; ws += (size_t)kTok * kD * sizeof(_Float16);
  _Float16* Ctx = (_Float16*)ws; ws += (size_t)kTok * kD * sizeof(_Float16);
  float*    Yf  = (float*)ws;

  const int nX = kTok * kD, nW = kD * kD;
  cvt_f32_f16<<<(nX + 255) / 256, 256, 0, stream>>>(ipt, Xh, nX);
  cvt_f32_f16<<<(nW + 255) / 256, 256, 0, stream>>>(Wq, Wqh, nW);
  cvt_f32_f16<<<(nW + 255) / 256, 256, 0, stream>>>(Wk, Wkh, nW);
  cvt_f32_f16<<<(nW + 255) / 256, 256, 0, stream>>>(Wv, Wvh, nW);
  cvt_f32_f16<<<(nW + 255) / 256, 256, 0, stream>>>(Wo, Woh, nW);

  dim3 gg(kD / 128, kTok / 128);
  gemm_f16_wmma<<<gg, 256, 0, stream>>>(Xh, Wqh, Qh, nullptr, nullptr, kTok, kD, kD);
  gemm_f16_wmma<<<gg, 256, 0, stream>>>(Xh, Wkh, Kh, nullptr, nullptr, kTok, kD, kD);
  gemm_f16_wmma<<<gg, 256, 0, stream>>>(Xh, Wvh, Vh, nullptr, nullptr, kTok, kD, kD);

  dim3 ga(kB * kH, kS / 128);
  flash_attn_wmma<<<ga, 256, 0, stream>>>(Qh, Kh, Vh, Ctx);

  gemm_f16_wmma<<<gg, 256, 0, stream>>>(Ctx, Woh, nullptr, Yf, ipt, kTok, kD, kD);
  layernorm_rows<<<kTok, 256, 0, stream>>>(Yf, out);
}
